// TemporalAttention_59957743452647
// MI455X (gfx1250) — matrix operand order repaired and checked by reading, not compiled, not run on hardware
//
#include <hip/hip_runtime.h>
#include <hip/hip_bf16.h>
#include <math.h>

// ---------------------------------------------------------------------------
// TemporalAttention for MI455X (gfx1250, wave32, WMMA bf16 16x16x32, f32 acc)
//   B=2 L=16 H=W=64 C=256  ->  N=B*L=32, HW=4096
// Pipeline (4 kernels):
//   1) qkv_proj  : Q/K/V = W*x(+b) as [131072 x 256]x[256 x 256] bf16 WMMA GEMMs
//                  Q  -> Qb[n][c][s]            (bf16, contiguous in s)
//                  K  -> Kb[n][d][t], t = spatial-transposed s (scatter on store)
//                  V  -> Vt[n][s][d] with torch's buggy-reshape index scramble
//                        applied at store time (so GEMM-2 A-loads are contiguous)
//   2) attn_logits: per n, [256 x 4096] x [4096 x 256]^T, 128 WMMA k-steps/tile
//   3) softmax    : row softmax over d (8192 rows x 256), bf16 probabilities
//   4) out_gemm   : per n, [4096 x 256] x [256 x 256], epilogue gamma*acc + x
// Workspace: Qb 64MB | Kb 64MB | Vt 64MB | logits f32 8MB | probs bf16 4MB
// bf16 conversion uses native fptrunc (v_cvt_*bf16*) instead of manual RNE
// bit-twiddling to keep the VALU pipe clear for WMMA co-execution.
// ---------------------------------------------------------------------------

typedef __attribute__((ext_vector_type(16))) __bf16        v16bf;
typedef __attribute__((ext_vector_type(8)))  float         v8f;
typedef __attribute__((ext_vector_type(4)))  float         f32x4;
typedef __attribute__((ext_vector_type(8)))  unsigned short u16x8;
typedef __attribute__((ext_vector_type(16))) unsigned short u16x16;

#define CC   256
#define HW   4096
#define NN   32

// float -> bf16 bits via native conversion
__device__ __forceinline__ unsigned short f2bf(float f) {
  return __builtin_bit_cast(unsigned short, (__bf16)f);
}

__device__ __forceinline__ v16bf pack16(u16x8 lo, u16x8 hi) {
  u16x16 t;
#pragma unroll
  for (int i = 0; i < 8; ++i) { t[i] = lo[i]; t[8 + i] = hi[i]; }
  return __builtin_bit_cast(v16bf, t);
}

__device__ __forceinline__ v16bf cvt16(f32x4 a, f32x4 b, f32x4 c, f32x4 d) {
  v16bf t;
#pragma unroll
  for (int i = 0; i < 4; ++i) {
    t[i]      = (__bf16)a[i];
    t[4 + i]  = (__bf16)b[i];
    t[8 + i]  = (__bf16)c[i];
    t[12 + i] = (__bf16)d[i];
  }
  return t;
}

__device__ __forceinline__ v8f wmma_bf16(v16bf a, v16bf b, v8f c) {
  // D = A(16x32 bf16) * B(32x16 bf16) + C(f32)
  return __builtin_amdgcn_wmma_f32_16x16x32_bf16(false, a, false, b,
                                                 (short)0, c, false, false);
}

// ---------------------------------------------------------------------------
// Kernel 1: fused Q/K/V projection. One wave = 16 rows x 256 cols of one
// projection; wave id -> (rowTile, proj). A operand read from fp32 inputs
// (channel-last => contiguous along K), B operand = W rows (contiguous K).
// ---------------------------------------------------------------------------
__global__ __launch_bounds__(256)
void qkv_proj_kernel(const float* __restrict__ X, const float* __restrict__ P,
                     const float* __restrict__ Wq, const float* __restrict__ bq,
                     const float* __restrict__ Wk, const float* __restrict__ bk,
                     const float* __restrict__ Wv, const float* __restrict__ bv,
                     unsigned short* __restrict__ Qb,
                     unsigned short* __restrict__ Kb,
                     unsigned short* __restrict__ Vt) {
  const int wid     = (blockIdx.x << 3) + (threadIdx.x >> 5);
  const int lane    = threadIdx.x & 31;
  const int proj    = wid % 3;
  const int rowTile = wid / 3;          // 0..8191
  const int rowBase = rowTile << 4;     // global row = (n, s) with s fast
  const int n       = rowBase >> 12;
  const int sB      = rowBase & (HW - 1);
  const int m       = lane & 15;
  const int kh      = lane >> 4;

  const float* Ain  = (proj == 0) ? X  : P;
  const float* Wm   = (proj == 0) ? Wq : (proj == 1 ? Wk : Wv);
  const float* bias = (proj == 0) ? bq : (proj == 1 ? bk : bv);

  const float* arow = Ain + (size_t)(rowBase + m) * CC;

#pragma unroll 1
  for (int ct = 0; ct < 16; ++ct) {
    const int c = (ct << 4) + m;                 // B column == output channel
    const float* bcol = Wm + (size_t)c * CC;     // W row c == B column c
    v8f acc = {};
#pragma unroll
    for (int kt = 0; kt < 8; ++kt) {
      const int k0 = (kt << 5) + (kh << 3);      // A: K groups {k0..k0+7, k0+16..k0+23}
      f32x4 a0 = *(const f32x4*)(arow + k0);
      f32x4 a1 = *(const f32x4*)(arow + k0 + 4);
      f32x4 a2 = *(const f32x4*)(arow + k0 + 16);
      f32x4 a3 = *(const f32x4*)(arow + k0 + 20);
      const int kb = (kt << 5) + (kh << 3);      // B: K groups {kb..kb+7, kb+16..kb+23}
      f32x4 b0 = *(const f32x4*)(bcol + kb);
      f32x4 b1 = *(const f32x4*)(bcol + kb + 4);
      f32x4 b2 = *(const f32x4*)(bcol + kb + 16);
      f32x4 b3 = *(const f32x4*)(bcol + kb + 20);
      acc = wmma_bf16(cvt16(a0, a1, a2, a3), cvt16(b0, b1, b2, b3), acc);
    }
    const float bz = bias[c];
    if (proj == 0) {
      // Q[n][c][s]: 8 consecutive s per lane -> one b128 store
      u16x8 st;
#pragma unroll
      for (int v = 0; v < 8; ++v) st[v] = f2bf(acc[v] + bz);
      *(u16x8*)(Qb + (((size_t)(n * CC + c)) << 12) + sB + (kh << 3)) = st;
    } else if (proj == 1) {
      // K[n][d][t] with t = (s%64)*64 + s/64 (torch's W,H swap before flatten)
      const size_t base = ((size_t)(n * CC + c)) << 12;
#pragma unroll
      for (int v = 0; v < 8; ++v) {
        const int s = sB + (kh << 3) + v;
        const int t = ((s & 63) << 6) | (s >> 6);
        Kb[base + t] = f2bf(acc[v] + bz);
      }
    } else {
      // Vt[nn][s][d]: buggy reshape of (B,Cout,L,HW) into (32,256,4096),
      // stored pre-transposed so the out-GEMM A-loads are contiguous in d.
      const int b  = n >> 4, lf = n & 15;
      const int nn = (b << 4) + (c >> 4);
      const int d  = ((c & 15) << 4) + lf;
#pragma unroll
      for (int v = 0; v < 8; ++v) {
        const int s = sB + (kh << 3) + v;
        Vt[((((size_t)nn << 12) + s) << 8) + d] = f2bf(acc[v] + bz);
      }
    }
  }
}

// ---------------------------------------------------------------------------
// Kernel 2: attention logits. One wave = one 16x16 tile of attn[n] (256x256),
// reduction K = HW = 4096 -> 128 WMMA steps. Q/K rows are contiguous bf16.
// ---------------------------------------------------------------------------
__global__ __launch_bounds__(256)
void attn_logits_kernel(const unsigned short* __restrict__ Qb,
                        const unsigned short* __restrict__ Kb,
                        float* __restrict__ logits) {
  const int wid  = (blockIdx.x << 3) + (threadIdx.x >> 5);
  const int lane = threadIdx.x & 31;
  const int dt = wid & 15, ct = (wid >> 4) & 15, n = wid >> 8;
  const int m = lane & 15, kh = lane >> 4;

  const unsigned short* qrow = Qb + ((size_t)(n * CC + (ct << 4) + m) << 12);
  const unsigned short* krow = Kb + ((size_t)(n * CC + (dt << 4) + m) << 12);

  v8f acc = {};
#pragma unroll 8
  for (int kt = 0; kt < 128; ++kt) {
    const int k0 = (kt << 5) + (kh << 3);
    __builtin_prefetch(qrow + k0 + 256, 0, 1);   // global_prefetch next tiles
    u16x8 a0 = *(const u16x8*)(qrow + k0);
    u16x8 a1 = *(const u16x8*)(qrow + k0 + 16);
    const int kb = (kt << 5) + (kh << 3);
    u16x8 b0 = *(const u16x8*)(krow + kb);
    u16x8 b1 = *(const u16x8*)(krow + kb + 16);
    acc = wmma_bf16(pack16(a0, a1), pack16(b0, b1), acc);
  }
  float* orow = logits + ((size_t)(n * CC + (ct << 4) + (kh << 3)) << 8)
                       + (dt << 4) + m;
#pragma unroll
  for (int v = 0; v < 8; ++v) orow[(size_t)v << 8] = acc[v];
}

// ---------------------------------------------------------------------------
// Kernel 3: row softmax over d. One wave per (n,c) row of 256, 8 elems/lane,
// wave32 shuffle reductions, bf16 output.
// ---------------------------------------------------------------------------
__global__ __launch_bounds__(256)
void softmax_kernel(const float* __restrict__ logits,
                    unsigned short* __restrict__ probs) {
  const int row  = (blockIdx.x << 3) + (threadIdx.x >> 5);   // 0..8191
  const int lane = threadIdx.x & 31;
  const float* r = logits + ((size_t)row << 8);
  float v[8];
  float mx = -3.4e38f;
#pragma unroll
  for (int i = 0; i < 8; ++i) { v[i] = r[lane + (i << 5)]; mx = fmaxf(mx, v[i]); }
#pragma unroll
  for (int off = 16; off > 0; off >>= 1) mx = fmaxf(mx, __shfl_xor(mx, off, 32));
  float s = 0.f;
#pragma unroll
  for (int i = 0; i < 8; ++i) { v[i] = __expf(v[i] - mx); s += v[i]; }
#pragma unroll
  for (int off = 16; off > 0; off >>= 1) s += __shfl_xor(s, off, 32);
  const float inv = 1.0f / s;
  unsigned short* o = probs + ((size_t)row << 8);
#pragma unroll
  for (int i = 0; i < 8; ++i) o[lane + (i << 5)] = f2bf(v[i] * inv);
}

// ---------------------------------------------------------------------------
// Kernel 4: out[n][s][c] = sum_d probs[n][c][d] * Vt[n][s][d]; epilogue
// gamma*acc + x.  A = Vt rows (contiguous d), B = probs rows (contiguous d).
// ---------------------------------------------------------------------------
__global__ __launch_bounds__(256)
void out_gemm_kernel(const unsigned short* __restrict__ Vt,
                     const unsigned short* __restrict__ Pp,
                     const float* __restrict__ X,
                     const float* __restrict__ gamma,
                     float* __restrict__ out) {
  const int wid  = (blockIdx.x << 3) + (threadIdx.x >> 5);
  const int lane = threadIdx.x & 31;
  const int ct = wid & 15, st = (wid >> 4) & 255, n = wid >> 12;
  const int m = lane & 15, kh = lane >> 4;
  const int sB = st << 4;

  const unsigned short* arow = Vt + ((size_t)((n << 12) + sB + m) << 8);
  const int c = (ct << 4) + m;
  const unsigned short* brow = Pp + ((size_t)(n * CC + c) << 8);

  v8f acc = {};
#pragma unroll
  for (int kt = 0; kt < 8; ++kt) {
    const int k0 = (kt << 5) + (kh << 3);
    u16x8 a0 = *(const u16x8*)(arow + k0);
    u16x8 a1 = *(const u16x8*)(arow + k0 + 16);
    const int kb = (kt << 5) + (kh << 3);
    u16x8 b0 = *(const u16x8*)(brow + kb);
    u16x8 b1 = *(const u16x8*)(brow + kb + 16);
    acc = wmma_bf16(pack16(a0, a1), pack16(b0, b1), acc);
  }
  const float g = gamma[0];
  const size_t base = ((size_t)((n << 12) + sB + (kh << 3)) << 8) + c;
#pragma unroll
  for (int v = 0; v < 8; ++v) {
    const size_t idx = base + ((size_t)v << 8);
    out[idx] = fmaf(g, acc[v], X[idx]);
  }
}

// ---------------------------------------------------------------------------
extern "C" void kernel_launch(void* const* d_in, const int* in_sizes, int n_in,
                              void* d_out, int out_size, void* d_ws, size_t ws_size,
                              hipStream_t stream) {
  const float* x     = (const float*)d_in[0];
  const float* prev  = (const float*)d_in[1];
  const float* Wq    = (const float*)d_in[2];
  const float* bq    = (const float*)d_in[3];
  const float* Wk    = (const float*)d_in[4];
  const float* bk    = (const float*)d_in[5];
  const float* Wv    = (const float*)d_in[6];
  const float* bv    = (const float*)d_in[7];
  const float* gamma = (const float*)d_in[8];
  float* out = (float*)d_out;

  char* ws = (char*)d_ws;
  const size_t MB = 1ull << 20;
  unsigned short* Qb = (unsigned short*)(ws);              //  64 MB
  unsigned short* Kb = (unsigned short*)(ws + 64 * MB);    //  64 MB
  unsigned short* Vt = (unsigned short*)(ws + 128 * MB);   //  64 MB
  float*          Lg = (float*)         (ws + 192 * MB);   //   8 MB
  unsigned short* Pp = (unsigned short*)(ws + 200 * MB);   //   4 MB

  // 8192 row-tiles x 3 projections = 24576 waves / 8 per block
  hipLaunchKernelGGL(qkv_proj_kernel, dim3(3072), dim3(256), 0, stream,
                     x, prev, Wq, bq, Wk, bk, Wv, bv, Qb, Kb, Vt);
  // 32 n x 16 x 16 tiles = 8192 waves
  hipLaunchKernelGGL(attn_logits_kernel, dim3(1024), dim3(256), 0, stream,
                     Qb, Kb, Lg);
  // 8192 rows
  hipLaunchKernelGGL(softmax_kernel, dim3(1024), dim3(256), 0, stream, Lg, Pp);
  // 32 n x 256 s-tiles x 16 c-tiles = 131072 waves
  hipLaunchKernelGGL(out_gemm_kernel, dim3(16384), dim3(256), 0, stream,
                     Vt, Pp, x, gamma, out);
}
